// CausalMultiHeadAttention_90460601189252
// MI455X (gfx1250) — matrix operand order repaired and checked by reading, not compiled, not run on hardware
//
#include <hip/hip_runtime.h>
#include <hip/hip_bf16.h>
#include <math.h>
#include <stdint.h>

#define SEQ    4096
#define DMODEL 1024
#define NHEADS 16
#define DK     64

typedef __attribute__((ext_vector_type(16))) _Float16 v16h;
typedef __attribute__((ext_vector_type(8)))  _Float16 v8h;
typedef __attribute__((ext_vector_type(8)))  float    v8f;

union AFrag { v16h v; v8h h[2]; };
union CFrag { v8f  v; float f[8]; };

// ---- WMMA wrapper: D = A(16x32 f16) * B(32x16 f16) + C(16x16 f32) ----
__device__ __forceinline__ v8f wmma_f32_f16(v16h a, v16h b, v8f c) {
  return __builtin_amdgcn_wmma_f32_16x16x32_f16(
      /*neg_a=*/false, a, /*neg_b=*/false, b,
      /*c_mod=*/(short)0, c, /*reuse_a=*/false, /*reuse_b=*/false);
}

// A fragment (16x32, 16-bit): lane m = lane&15; lanes<16 hold K 0..15,
// lanes>=16 hold K 16..31 (contiguous).  base = &A[m0*lda + k0]
__device__ __forceinline__ v16h load_a_frag(const _Float16* __restrict__ base,
                                            int lda, int lane) {
  const int m  = lane & 15;
  const int kb = (lane >> 4) << 4;  // 0 or 16
  AFrag f;
  f.h[0] = *(const v8h*)(base + m * lda + kb);
  f.h[1] = *(const v8h*)(base + m * lda + kb + 8);
  return f.v;
}

// B fragment (32x16, 16-bit), B stored as N rows x K cols (i.e. W[n][k]):
// lane n = lane&15; lanes<16 hold K 0..15, lanes>=16 hold K 16..31 (contiguous).
__device__ __forceinline__ v16h load_b_frag(const _Float16* __restrict__ base,
                                            int ldb, int lane) {
  const int n  = lane & 15;
  const int kb = (lane >> 4) << 4;  // 0 or 16
  return *(const v16h*)(base + n * ldb + kb);
}

__device__ __forceinline__ float redmax16(float x) {
#pragma unroll
  for (int m = 1; m < 16; m <<= 1) x = fmaxf(x, __shfl_xor(x, m, 32));
  return x;
}
__device__ __forceinline__ float redsum16(float x) {
#pragma unroll
  for (int m = 1; m < 16; m <<= 1) x += __shfl_xor(x, m, 32);
  return x;
}

// ---- CDNA5 async copy global->LDS (ASYNCcnt-tracked), 16B per lane ----
__device__ __forceinline__ void async_ld_b128(uint32_t lds_off,
                                              const void* gptr) {
  asm volatile("global_load_async_to_lds_b128 %0, %1, off"
               :: "v"(lds_off), "v"(gptr) : "memory");
}
__device__ __forceinline__ void wait_async_le(int n) {
  if (n == 0) {
#if __has_builtin(__builtin_amdgcn_s_wait_asynccnt)
    __builtin_amdgcn_s_wait_asynccnt(0);
#else
    asm volatile("s_wait_asynccnt 0x0" ::: "memory");
#endif
  } else {
#if __has_builtin(__builtin_amdgcn_s_wait_asynccnt)
    __builtin_amdgcn_s_wait_asynccnt(4);
#else
    asm volatile("s_wait_asynccnt 0x4" ::: "memory");
#endif
  }
}
__device__ __forceinline__ uint32_t lds_off_of(const void* p) {
  // Flat aperture mapping: LDS offset is the low 32 bits of the address  (ISA 10.2)
  return (uint32_t)(uintptr_t)p;
}

// ---------------------------------------------------------------- converts
__global__ void cvt_f32_f16_kernel(const float* __restrict__ src,
                                   _Float16* __restrict__ dst, int n) {
  int i = blockIdx.x * blockDim.x + threadIdx.x;
  if (i < n) dst[i] = (_Float16)src[i];
}

// --------------------------------------------------- fused QKV GEMM + RoPE
// 32x64 tile per wave: 2 A frags x 4 B frags -> 8 wmma per 32-k step
// z=0: Q -> rope -> Qh[h][s][dk]; z=1: K -> rope -> Kh[h][s][dk]
// z=2: V -> Vt[h][dk][s]
__global__ __launch_bounds__(256) void qkv_rope_kernel(
    const _Float16* __restrict__ Xh, const int* __restrict__ pos,
    const _Float16* __restrict__ Wq, const _Float16* __restrict__ Wk,
    const _Float16* __restrict__ Wv,
    _Float16* __restrict__ Qh, _Float16* __restrict__ Kh,
    _Float16* __restrict__ Vt) {
  const int z = blockIdx.y;
  const _Float16* __restrict__ W = (z == 0) ? Wq : ((z == 1) ? Wk : Wv);
  const int lane = threadIdx.x & 31;
  const int wave = threadIdx.x >> 5;
  const int gw = blockIdx.x * 8 + wave;   // 2048 wave-tiles
  const int m0 = (gw >> 4) << 5;          // 128 row tiles of 32
  const int n0 = (gw & 15) << 6;          // 16 col strips of 64

  CFrag acc[2][4];
#pragma unroll
  for (int mt = 0; mt < 2; ++mt)
#pragma unroll
    for (int t = 0; t < 4; ++t) acc[mt][t].v = v8f{};

  for (int k0 = 0; k0 < DMODEL; k0 += 32) {
    __builtin_prefetch(W + n0 * DMODEL + k0 + 64, 0, 1);  // global_prefetch_b8
    v16h a0 = load_a_frag(Xh + m0 * DMODEL + k0, DMODEL, lane);
    v16h a1 = load_a_frag(Xh + (m0 + 16) * DMODEL + k0, DMODEL, lane);
#pragma unroll
    for (int t = 0; t < 4; ++t) {
      v16h b = load_b_frag(W + (n0 + 16 * t) * DMODEL + k0, DMODEL, lane);
      acc[0][t].v = wmma_f32_f16(a0, b, acc[0][t].v);
      acc[1][t].v = wmma_f32_f16(a1, b, acc[1][t].v);
    }
  }

  const int rbase = (lane >> 4) << 3;
  const int nlane = lane & 15;
#pragma unroll
  for (int t = 0; t < 4; ++t) {
    const int n = n0 + 16 * t + nlane;
    const int h = n >> 6;
    const int d = n & 63;
    if (z < 2) {
      const float inv =
          __expf(-(float)(d & ~1) * (9.210340371976184f / 64.0f));
      _Float16* __restrict__ out = (z == 0) ? Qh : Kh;
#pragma unroll
      for (int mt = 0; mt < 2; ++mt)
#pragma unroll
        for (int r = 0; r < 8; ++r) {
          const int m = m0 + 16 * mt + rbase + r;
          const float ang = (float)pos[m] * inv;
          const float c = __cosf(ang);
          const float s = __sinf(ang);
          const float v = acc[mt][t].f[r];
          const float p = __shfl_xor(v, 1, 32);
          const float o = (d & 1) ? (p * s + v * c) : (v * c - p * s);
          out[(h * SEQ + m) * DK + d] = (_Float16)o;
        }
    } else {
#pragma unroll
      for (int mt = 0; mt < 2; ++mt)
#pragma unroll
        for (int r = 0; r < 8; ++r) {
          const int m = m0 + 16 * mt + rbase + r;
          Vt[(h * DK + d) * SEQ + m] = (_Float16)acc[mt][t].f[r];
        }
    }
  }
}

// ------------------------------------------------- causal flash attention
// Block = 4 waves, one 16-query tile each, same head.  K/V 32-key chunks are
// staged in LDS once per block via async-to-LDS, double buffered; chunk c+1
// copies overlap chunk c's 8 WMMAs.
__global__ __launch_bounds__(128) void attn_kernel(
    const _Float16* __restrict__ Qh, const _Float16* __restrict__ Kh,
    const _Float16* __restrict__ Vt, _Float16* __restrict__ Ah) {
  __shared__ __align__(32) _Float16 Klds[2][32][DK];  // [key][d]
  __shared__ __align__(32) _Float16 Vlds[2][DK][32];  // [d][key]
  __shared__ __align__(32) _Float16 Pbuf[4][16][32];

  const int tid = threadIdx.x;
  const int lane = tid & 31;
  const int wave = tid >> 5;
  const int h = blockIdx.y;
  const int qt = blockIdx.x * 4 + wave;
  const int q0 = qt * 16;
  const int rbase = (lane >> 4) << 3;
  const int nlane = lane & 15;

  const int my_nch  = (q0 + 16 + 31) >> 5;
  const int max_nch = ((blockIdx.x * 4 + 3) * 16 + 16 + 31) >> 5;

  const _Float16* __restrict__ Qbase = Qh + (h * SEQ + q0) * DK;
  const v16h qa0 = load_a_frag(Qbase + 0, DK, lane);
  const v16h qa1 = load_a_frag(Qbase + 32, DK, lane);

  CFrag o[4];
#pragma unroll
  for (int t = 0; t < 4; ++t) o[t].v = v8f{};
  float rmax[8], rsum[8];
#pragma unroll
  for (int r = 0; r < 8; ++r) { rmax[r] = -1.0e30f; rsum[r] = 0.0f; }

  // cooperative async stage of one 32-key chunk (4 async b128 per thread)
  auto issue = [&](int c) {
    const int buf = c & 1;
    const int kc0 = c << 5;
    {  // K chunk: 32 keys x 64 d = 4KB contiguous
      const char* src = (const char*)(Kh + (h * SEQ + kc0) * DK);
      const uint32_t dst = lds_off_of(&Klds[buf][0][0]);
      async_ld_b128(dst + tid * 16, src + tid * 16);
      async_ld_b128(dst + 2048 + tid * 16, src + 2048 + tid * 16);
    }
    {  // V chunk: 64 d-rows x 32 keys (64B rows, stride SEQ in Vt)
      const uint32_t dst = lds_off_of(&Vlds[buf][0][0]);
#pragma unroll
      for (int j = 0; j < 2; ++j) {
        const int p = tid + 128 * j;      // 0..255 16B pieces
        const int row = p >> 2, seg = p & 3;
        async_ld_b128(dst + (row * 32 + seg * 8) * 2,
                      Vt + ((h * DK + row) * SEQ + kc0 + seg * 8));
      }
    }
  };

  issue(0);
  for (int c = 0; c < max_nch; ++c) {
    const int buf = c & 1;
    if (c + 1 < max_nch) {
      issue(c + 1);
      wait_async_le(4);   // chunk c's 4 copies done; c+1's may be in flight
    } else {
      wait_async_le(0);
    }
    __syncthreads();      // everyone's copies for chunk c visible

    if (c < my_nch) {     // wave-uniform guard (EXEC all-ones for WMMA)
      const int kc0 = c << 5;
      const int kb16 = (lane >> 4) << 4;
      // ---- S = Q * K^T, two 16-key subtiles from LDS
      CFrag s[2];
#pragma unroll
      for (int t = 0; t < 2; ++t) {
        const int j = 16 * t + nlane;
        v16h b0 = *(const v16h*)&Klds[buf][j][kb16];
        v16h b1 = *(const v16h*)&Klds[buf][j][kb16 + 32];
        v8f zz = v8f{};
        zz = wmma_f32_f16(qa0, b0, zz);
        zz = wmma_f32_f16(qa1, b1, zz);
        s[t].v = zz;
      }
      // ---- scale, causal mask, online softmax
      const bool need_mask = (kc0 + 31 > q0);
      float p0[8], p1[8], alpha[8];
#pragma unroll
      for (int r = 0; r < 8; ++r) {
        const int qrow = q0 + rbase + r;
        float v0 = s[0].f[r] * 0.125f;
        float v1 = s[1].f[r] * 0.125f;
        if (need_mask) {
          if (kc0 + nlane > qrow)      v0 = -1.0e30f;
          if (kc0 + 16 + nlane > qrow) v1 = -1.0e30f;
        }
        const float cmax = redmax16(fmaxf(v0, v1));
        const float nmax = fmaxf(rmax[r], cmax);
        alpha[r] = __expf(rmax[r] - nmax);
        rmax[r] = nmax;
        const float e0 = __expf(v0 - nmax);
        const float e1 = __expf(v1 - nmax);
        p0[r] = e0;
        p1[r] = e1;
        rsum[r] = rsum[r] * alpha[r] + redsum16(e0 + e1);
      }
#pragma unroll
      for (int t = 0; t < 4; ++t)
#pragma unroll
        for (int r = 0; r < 8; ++r) o[t].f[r] *= alpha[r];

      // ---- repack P: C layout (col/lane) -> A layout (row/lane) via LDS
#pragma unroll
      for (int r = 0; r < 8; ++r) {
        Pbuf[wave][rbase + r][nlane]      = (_Float16)p0[r];
        Pbuf[wave][rbase + r][16 + nlane] = (_Float16)p1[r];
      }
      AFrag pf;
      {
        const int m = lane & 15;
        const int kb = (lane >> 4) << 4;
        pf.h[0] = *(const v8h*)&Pbuf[wave][m][kb];
        pf.h[1] = *(const v8h*)&Pbuf[wave][m][kb + 8];
      }
      // ---- O += P * V from LDS (column d contiguous along keys)
#pragma unroll
      for (int t = 0; t < 4; ++t) {
        v16h vb = *(const v16h*)&Vlds[buf][16 * t + nlane][kb16];
        o[t].v = wmma_f32_f16(pf.v, vb, o[t].v);
      }
    }
    __syncthreads();      // compute done before buffer is overwritten
  }

  // ---- epilogue: normalize + store f16 [s][dmodel]
#pragma unroll
  for (int t = 0; t < 4; ++t)
#pragma unroll
    for (int r = 0; r < 8; ++r) {
      const int m = q0 + rbase + r;
      const float val = o[t].f[r] / rsum[r];
      Ah[m * DMODEL + h * DK + 16 * t + nlane] = (_Float16)val;
    }
}

// ------------------------------------------------------ output projection
__global__ __launch_bounds__(256) void oproj_kernel(
    const _Float16* __restrict__ Ah, const _Float16* __restrict__ Wo,
    float* __restrict__ out) {
  const int lane = threadIdx.x & 31;
  const int wave = threadIdx.x >> 5;
  const int gw = blockIdx.x * 8 + wave;
  const int m0 = (gw >> 4) << 5;
  const int n0 = (gw & 15) << 6;

  CFrag acc[2][4];
#pragma unroll
  for (int mt = 0; mt < 2; ++mt)
#pragma unroll
    for (int t = 0; t < 4; ++t) acc[mt][t].v = v8f{};

  for (int k0 = 0; k0 < DMODEL; k0 += 32) {
    __builtin_prefetch(Wo + n0 * DMODEL + k0 + 64, 0, 1);
    v16h a0 = load_a_frag(Ah + m0 * DMODEL + k0, DMODEL, lane);
    v16h a1 = load_a_frag(Ah + (m0 + 16) * DMODEL + k0, DMODEL, lane);
#pragma unroll
    for (int t = 0; t < 4; ++t) {
      v16h b = load_b_frag(Wo + (n0 + 16 * t) * DMODEL + k0, DMODEL, lane);
      acc[0][t].v = wmma_f32_f16(a0, b, acc[0][t].v);
      acc[1][t].v = wmma_f32_f16(a1, b, acc[1][t].v);
    }
  }

  const int rbase = (lane >> 4) << 3;
  const int nlane = lane & 15;
#pragma unroll
  for (int mt = 0; mt < 2; ++mt)
#pragma unroll
    for (int t = 0; t < 4; ++t)
#pragma unroll
      for (int r = 0; r < 8; ++r) {
        const int m = m0 + 16 * mt + rbase + r;
        out[m * DMODEL + n0 + 16 * t + nlane] = acc[mt][t].f[r];
      }
}

extern "C" void kernel_launch(void* const* d_in, const int* in_sizes, int n_in,
                              void* d_out, int out_size, void* d_ws,
                              size_t ws_size, hipStream_t stream) {
  (void)in_sizes; (void)n_in; (void)out_size; (void)ws_size;
  const float* x  = (const float*)d_in[0];
  const int*  pos = (const int*)d_in[1];
  const float* Wq = (const float*)d_in[2];
  const float* Wk = (const float*)d_in[3];
  const float* Wv = (const float*)d_in[4];
  const float* Wv2 = (const float*)d_in[4];
  const float* Wo = (const float*)d_in[5];
  (void)Wv2;
  float* out = (float*)d_out;

  _Float16* ws = (_Float16*)d_ws;
  size_t off = 0;
  _Float16* Xh  = ws + off; off += (size_t)SEQ * DMODEL;
  _Float16* Wqh = ws + off; off += (size_t)DMODEL * DMODEL;
  _Float16* Wkh = ws + off; off += (size_t)DMODEL * DMODEL;
  _Float16* Wvh = ws + off; off += (size_t)DMODEL * DMODEL;
  _Float16* Woh = ws + off; off += (size_t)DMODEL * DMODEL;
  _Float16* Qh  = ws + off; off += (size_t)SEQ * DMODEL;
  _Float16* Kh  = ws + off; off += (size_t)SEQ * DMODEL;
  _Float16* Vt  = ws + off; off += (size_t)SEQ * DMODEL;
  _Float16* Ah  = ws + off; off += (size_t)SEQ * DMODEL;

  const int nX = SEQ * DMODEL;
  const int nW = DMODEL * DMODEL;
  cvt_f32_f16_kernel<<<(nX + 255) / 256, 256, 0, stream>>>(x, Xh, nX);
  cvt_f32_f16_kernel<<<(nW + 255) / 256, 256, 0, stream>>>(Wq, Wqh, nW);
  cvt_f32_f16_kernel<<<(nW + 255) / 256, 256, 0, stream>>>(Wk, Wkh, nW);
  cvt_f32_f16_kernel<<<(nW + 255) / 256, 256, 0, stream>>>(Wv, Wvh, nW);
  cvt_f32_f16_kernel<<<(nW + 255) / 256, 256, 0, stream>>>(Wo, Woh, nW);

  qkv_rope_kernel<<<dim3(256, 3), 256, 0, stream>>>(Xh, pos, Wqh, Wkh, Wvh,
                                                    Qh, Kh, Vt);
  attn_kernel<<<dim3(64, 16), 128, 0, stream>>>(Qh, Kh, Vt, Ah);
  oproj_kernel<<<dim3(256, 1), 256, 0, stream>>>(Ah, Woh, out);
}
